// SimpleVDforPreGate_88029649699060
// MI455X (gfx1250) — compile-verified, matrix operand order checked by reading, not run on hardware
//
#include <hip/hip_runtime.h>
#include <hip/hip_bf16.h>

// ---------------------------------------------------------------------------
// Types for CDNA5 WMMA (wave32): v16bf = 16xbf16 A/B fragment (8 VGPRs),
// v8f = 16x16 f32 C/D fragment (8 VGPRs).
// ---------------------------------------------------------------------------
typedef __attribute__((ext_vector_type(16))) __bf16        v16bf;
typedef __attribute__((ext_vector_type(8)))  float         v8f;
typedef __attribute__((ext_vector_type(4)))  unsigned int  uint4v;
typedef __attribute__((ext_vector_type(4)))  int           int4v;
// address-space-qualified int4 element types for the async-to-LDS builtin
typedef __attribute__((address_space(1))) int4v g_int4;    // global
typedef __attribute__((address_space(3))) int4v l_int4;    // LDS

#define TWO_PI_F 6.283185307179586f
#define LN10000_F 9.210340371976184f

#if __has_builtin(__builtin_amdgcn_global_load_async_to_lds_b128)
#define HAVE_ASYNC_LDS 1
#else
#define HAVE_ASYNC_LDS 0
#endif

__device__ __forceinline__ void wait_asynccnt0() {
#if __has_builtin(__builtin_amdgcn_s_wait_asynccnt)
  __builtin_amdgcn_s_wait_asynccnt(0);
#else
  asm volatile("s_wait_asynccnt 0x0" ::: "memory");
#endif
}

__device__ __forceinline__ unsigned short f32_to_bf16(float f) {
  unsigned int u = __float_as_uint(f);
  u += 0x7fffu + ((u >> 16) & 1u);   // round-to-nearest-even
  return (unsigned short)(u >> 16);
}

// Load one 16-bit A/B fragment: per-lane 8 contiguous bf16 at K=kb..kb+7 and
// 8 at K=kb+16..kb+23 (ISA 7.12.2 16-bit 16x32 layout).  p = rowBase + k + kb.
__device__ __forceinline__ v16bf frag_ld(const unsigned short* p) {
  union { uint4v q[2]; v16bf v; } u;
  u.q[0] = *(const uint4v*)(p);
  u.q[1] = *(const uint4v*)(p + 16);
  return u.v;
}

__device__ __forceinline__ v8f wmma_bf16(v16bf a, v16bf b, v8f c) {
  return __builtin_amdgcn_wmma_f32_16x16x32_bf16(
      /*neg_a=*/false, a, /*neg_b=*/false, b,
      /*c_mod=*/(short)0, c, /*reuse_a=*/false, /*reuse_b=*/false);
}

// ---------------------------------------------------------------------------
// Kernel 1: 2x2 maxpool of img (B,Cin,64,64) -> token-major bf16 A matrix
// x_t[t][c], t = b*1024 + h*32 + w.  Raw tile staged via async-to-LDS (when
// available), pooled, then LDS-transposed for coalesced bf16 writes.
// grid = B*H*(Cin/64) = 16*32*32, block = 256.
// ---------------------------------------------------------------------------
__global__ void maxpool_pack_kernel(const float* __restrict__ img,
                                    unsigned short* __restrict__ xt) {
  __shared__ float sm[64][33];
  const int cg  = blockIdx.x & 31;           // channel group (64 ch)
  const int h   = (blockIdx.x >> 5) & 31;    // pooled row
  const int b   = blockIdx.x >> 10;          // batch
  const int tid = threadIdx.x;

#if HAVE_ASYNC_LDS
  // stage 64 channels x 2 rows x 64 px (32KB) via GLOBAL_LOAD_ASYNC_TO_LDS
  __shared__ float raw[64 * 2 * 64];
  for (int i = 0; i < 8; ++i) {
    const int q  = tid + 256 * i;            // 16-byte chunk id, 2048 total
    const int c  = q >> 5;                   // 32 chunks per channel
    const int r  = (q >> 4) & 1;
    const int wq = (q & 15) * 4;
    const float* gp =
        img + (((size_t)(b * 2048 + cg * 64 + c) * 64 + 2 * h + r) * 64 + wq);
    float* lp = &raw[q * 4];
    __builtin_amdgcn_global_load_async_to_lds_b128(
        (g_int4*)gp, (l_int4*)lp, 0, 0);
  }
  wait_asynccnt0();
  __syncthreads();
  // pool from LDS
  for (int pass = 0; pass < 8; ++pass) {
    const int cl = pass * 8 + (tid >> 5);
    const int w  = tid & 31;
    const float* p = &raw[cl * 128 + 2 * w];
    sm[cl][w] = fmaxf(fmaxf(p[0], p[1]), fmaxf(p[64], p[65]));
  }
#else
  const int w = tid & 31;
  for (int pass = 0; pass < 8; ++pass) {
    const int cl = pass * 8 + (tid >> 5);
    const int c  = cg * 64 + cl;
    const float* p = img + (((size_t)(b * 2048 + c) * 64 + 2 * h) * 64 + 2 * w);
    sm[cl][w] = fmaxf(fmaxf(p[0], p[1]), fmaxf(p[64], p[65]));
  }
#endif
  __syncthreads();
  // write bf16, 64 consecutive channels per token (128B coalesced)
  const int cl = tid & 63;
  const int tb = tid >> 6;
  for (int i = 0; i < 8; ++i) {
    const int tok = tb + 4 * i;                        // w index 0..31
    const size_t t = (size_t)b * 1024 + h * 32 + tok;
    xt[t * 2048 + cg * 64 + cl] = f32_to_bf16(sm[cl][tok]);
  }
}

// ---------------------------------------------------------------------------
// Kernel 2: elementwise f32 -> bf16
// ---------------------------------------------------------------------------
__global__ void cvt_bf16_kernel(const float* __restrict__ in,
                                unsigned short* __restrict__ out, int n) {
  int i = blockIdx.x * blockDim.x + threadIdx.x;
  if (i < n) out[i] = f32_to_bf16(in[i]);
}

// ---------------------------------------------------------------------------
// Kernel 3: hn[k] = 0.5 * ||codebook[k]||^2   (one wave per row)
// ---------------------------------------------------------------------------
__global__ void halfnorm_kernel(const float* __restrict__ cb,
                                float* __restrict__ hn) {
  const int lane = threadIdx.x & 31;
  const int row  = blockIdx.x * 8 + (threadIdx.x >> 5);
  const float* p = cb + (size_t)row * 768;
  float s = 0.f;
  for (int c = lane; c < 768; c += 32) { float v = p[c]; s += v * v; }
  for (int off = 16; off >= 1; off >>= 1) s += __shfl_xor(s, off, 32);
  if (lane == 0) hn[row] = 0.5f * s;
}

// ---------------------------------------------------------------------------
// Kernel 4: WMMA GEMM  out[m][n] = sum_k A[m][k]*B[n][k] + bias[n]
// A rows optionally gathered through gatherIdx (quant = codebook[idx]).
// Block = 256 thr = 8 waves in 2(M) x 4(N); block tile 64x256, wave tile
// 32x64.  Each K-step: load all 6 fragments, then 8 WMMAs as one group so
// the scheduler overlaps next-step loads under the WMMA block.
// grid = (M/64, N/256).
// ---------------------------------------------------------------------------
__global__ void wmma_gemm_kernel(const unsigned short* __restrict__ A,
                                 const int* __restrict__ gatherIdx,
                                 const unsigned short* __restrict__ B,
                                 const float* __restrict__ bias,
                                 float* __restrict__ outF,
                                 unsigned short* __restrict__ outBF,
                                 int N, int K) {
  const int lane = threadIdx.x & 31;
  const int wave = threadIdx.x >> 5;
  const int wm = wave >> 2, wn = wave & 3;
  const int m0 = blockIdx.x * 64 + wm * 32;
  const int n0 = blockIdx.y * 256 + wn * 64;
  const int cr = lane & 15;
  const int kb = (lane >> 4) << 3;
  const int half = lane >> 4;

  const int rA0 = m0 + cr, rA1 = m0 + 16 + cr;
  const unsigned short* aRow0 =
      gatherIdx ? (A + (size_t)gatherIdx[rA0] * K) : (A + (size_t)rA0 * K);
  const unsigned short* aRow1 =
      gatherIdx ? (A + (size_t)gatherIdx[rA1] * K) : (A + (size_t)rA1 * K);
  const unsigned short* bRow[4];
  for (int ni = 0; ni < 4; ++ni)
    bRow[ni] = B + (size_t)(n0 + ni * 16 + cr) * K;

  v8f acc[2][4];
  for (int mi = 0; mi < 2; ++mi)
    for (int ni = 0; ni < 4; ++ni) acc[mi][ni] = (v8f){};

  for (int k = 0; k < K; k += 32) {
    __builtin_prefetch(aRow0 + k + 256, 0, 1);    // global_prefetch_b8
    // load all fragments first ...
    const v16bf a0 = frag_ld(aRow0 + k + kb);
    const v16bf a1 = frag_ld(aRow1 + k + kb);
    v16bf bf4[4];
#pragma unroll
    for (int ni = 0; ni < 4; ++ni) bf4[ni] = frag_ld(bRow[ni] + k + kb);
    // ... then issue the 8 WMMAs as one block
#pragma unroll
    for (int ni = 0; ni < 4; ++ni) {
      acc[0][ni] = wmma_bf16(a0, bf4[ni], acc[0][ni]);
      acc[1][ni] = wmma_bf16(a1, bf4[ni], acc[1][ni]);
    }
  }
#pragma unroll
  for (int ni = 0; ni < 4; ++ni) {
    const int c = n0 + ni * 16 + cr;
    const float bb = bias[c];
#pragma unroll
    for (int mi = 0; mi < 2; ++mi) {
      for (int i = 0; i < 8; ++i) {
        const int r = m0 + mi * 16 + i + 8 * half;   // C layout: VGPR i, half
        const float v = acc[mi][ni][i] + bb;
        outF[(size_t)r * N + c] = v;
        if (outBF) outBF[(size_t)r * N + c] = f32_to_bf16(v);
      }
    }
  }
}

// ---------------------------------------------------------------------------
// Kernel 5: VQ argmin via WMMA: score = f.c - 0.5||c||^2, argmax over 2048
// codes per token.  Block = 64 tokens, waves 2(M) x 4(N), wave tile 32x64,
// N looped in 256-code chunks.  Running argmax per C-fragment row; shfl-xor
// reduce over the 16-lane half; 4-way cross-wave combine in LDS.
// grid = M/64, block = 256.
// ---------------------------------------------------------------------------
__global__ void vq_argmax_kernel(const unsigned short* __restrict__ A,
                                 const unsigned short* __restrict__ CB,
                                 const float* __restrict__ hn,
                                 int* __restrict__ idxOut, int K, int NC) {
  const int lane = threadIdx.x & 31;
  const int wave = threadIdx.x >> 5;
  const int wm = wave >> 2, wn = wave & 3;
  const int m0 = blockIdx.x * 64 + wm * 32;
  const int cr = lane & 15;
  const int kb = (lane >> 4) << 3;
  const int half = lane >> 4;

  const unsigned short* aRow0 = A + (size_t)(m0 + cr) * K;
  const unsigned short* aRow1 = A + (size_t)(m0 + 16 + cr) * K;

  float bestv[2][8]; int besti[2][8];
  for (int mi = 0; mi < 2; ++mi)
    for (int i = 0; i < 8; ++i) { bestv[mi][i] = -3.0e38f; besti[mi][i] = 0; }

  for (int nb = 0; nb < NC; nb += 256) {
    const int n0 = nb + wn * 64;
    const unsigned short* bRow[4];
    float h4[4];
    for (int ni = 0; ni < 4; ++ni) {
      bRow[ni] = CB + (size_t)(n0 + ni * 16 + cr) * K;
      h4[ni] = hn[n0 + ni * 16 + cr];
    }
    v8f acc[2][4];
    for (int mi = 0; mi < 2; ++mi)
      for (int ni = 0; ni < 4; ++ni) acc[mi][ni] = (v8f){};
    for (int k = 0; k < K; k += 32) {
      const v16bf a0 = frag_ld(aRow0 + k + kb);
      const v16bf a1 = frag_ld(aRow1 + k + kb);
      v16bf bf4[4];
#pragma unroll
      for (int ni = 0; ni < 4; ++ni) bf4[ni] = frag_ld(bRow[ni] + k + kb);
#pragma unroll
      for (int ni = 0; ni < 4; ++ni) {
        acc[0][ni] = wmma_bf16(a0, bf4[ni], acc[0][ni]);
        acc[1][ni] = wmma_bf16(a1, bf4[ni], acc[1][ni]);
      }
    }
#pragma unroll
    for (int mi = 0; mi < 2; ++mi) {
      for (int i = 0; i < 8; ++i) {
        float sc = acc[mi][0][i] - h4[0]; int ci = n0 + cr;
#pragma unroll
        for (int ni = 1; ni < 4; ++ni) {
          const float s2 = acc[mi][ni][i] - h4[ni];
          const int   c2 = n0 + ni * 16 + cr;
          if (s2 > sc) { sc = s2; ci = c2; }        // larger ni => larger idx
        }
        // reduce across the 16 lanes of this half (xor keeps halves intact)
        for (int off = 1; off < 16; off <<= 1) {
          const float ov = __shfl_xor(sc, off, 32);
          const int   oi = __shfl_xor(ci, off, 32);
          if (ov > sc || (ov == sc && oi < ci)) { sc = ov; ci = oi; }
        }
        if (sc > bestv[mi][i] ||
            (sc == bestv[mi][i] && ci < besti[mi][i])) {
          bestv[mi][i] = sc; besti[mi][i] = ci;
        }
      }
    }
  }

  __shared__ float s_bv[4][64];
  __shared__ int   s_bi[4][64];
  if (cr == 0) {
    for (int mi = 0; mi < 2; ++mi)
      for (int i = 0; i < 8; ++i) {
        const int tl = wm * 32 + mi * 16 + i + 8 * half;
        s_bv[wn][tl] = bestv[mi][i];
        s_bi[wn][tl] = besti[mi][i];
      }
  }
  __syncthreads();
  if (threadIdx.x < 64) {
    float bv = s_bv[0][threadIdx.x];
    int   bi = s_bi[0][threadIdx.x];
    for (int q = 1; q < 4; ++q) {
      const float v = s_bv[q][threadIdx.x];
      const int   i = s_bi[q][threadIdx.x];
      if (v > bv || (v == bv && i < bi)) { bv = v; bi = i; }
    }
    idxOut[blockIdx.x * 64 + threadIdx.x] = bi;
  }
}

// ---------------------------------------------------------------------------
// Kernel 6: fused epilogue per token: gate softmax, blend, positional sincos,
// mask substitution, LayerNorm; also emits vm/labels (as float values).
// grid = 16384 (one token per block), block = 256 (3 channels/thread).
// ---------------------------------------------------------------------------
__global__ void epilogue_kernel(const float* __restrict__ emb,
                                const float* __restrict__ xq,
                                const int* __restrict__ idx,
                                const int* __restrict__ vis_hw,
                                const int* __restrict__ tmp_pos,
                                const float* __restrict__ bern,
                                const float* __restrict__ gate_w,
                                const float* __restrict__ gate_b,
                                const float* __restrict__ mask_emb,
                                const float* __restrict__ ln_g,
                                const float* __restrict__ ln_b,
                                float* __restrict__ out,
                                float* __restrict__ vmOut,
                                float* __restrict__ labOut) {
  __shared__ float r0[256], r1[256];
  const int t = blockIdx.x;
  const int b = t >> 10, hw = t & 1023, h = hw >> 5, w = hw & 31;
  const int tid = threadIdx.x;

  const int imh = vis_hw[2 * b], imw = vis_hw[2 * b + 1];
  const bool rh = (h * 32) < imh, rw = (w * 32) < imw;
  const float vmv = (rh && rw) ? 1.f : 0.f;
  const int cntH = (imh + 31) >> 5, cntW = (imw + 31) >> 5;
  const int cnth = (h + 1 < cntH) ? (h + 1) : cntH;
  const int cntw = (w + 1 < cntW) ? (w + 1) : cntW;
  const float yv = rw ? ((float)cnth / ((float)cntH + 1e-6f)) * TWO_PI_F : 0.f;
  const float xv = rh ? ((float)cntw / ((float)cntW + 1e-6f)) * TWO_PI_F : 0.f;

  const int tp = tmp_pos[0];
  const int th = tp >> 5, tw = tp & 31;
  const bool tvm = ((th * 32) < imh) && ((tw * 32) < imw);
  const float tl = tvm ? (float)idx[b * 1024 + tp] : -100.f;
  const float indv = (vmv > 0.f) ? (float)idx[t] : -100.f;
  const float mi = ((indv == tl) ? 1.f : 0.f) * vmv * bern[b];

  // gate logits (two 1536-length dot products, block-reduced)
  float ev[3], xv3[3];
  float p0 = 0.f, p1 = 0.f;
  for (int j = 0; j < 3; ++j) {
    const int c = tid + 256 * j;
    ev[j]  = emb[(size_t)t * 768 + c];
    xv3[j] = xq [(size_t)t * 768 + c];
    p0 += gate_w[c] * ev[j] + gate_w[768 + c] * xv3[j];
    p1 += gate_w[1536 + c] * ev[j] + gate_w[1536 + 768 + c] * xv3[j];
  }
  r0[tid] = p0; r1[tid] = p1; __syncthreads();
  for (int s = 128; s > 0; s >>= 1) {
    if (tid < s) { r0[tid] += r0[tid + s]; r1[tid] += r1[tid + s]; }
    __syncthreads();
  }
  const float s0 = r0[0] + gate_b[0], s1 = r1[0] + gate_b[1];
  const float mx = fmaxf(s0, s1);
  const float e0 = __expf(s0 - mx), e1 = __expf(s1 - mx);
  const float g0 = e0 / (e0 + e1), g1 = e1 / (e0 + e1);

  // fused + mask + positional
  float vals[3];
  float sum = 0.f, sq = 0.f;
  for (int j = 0; j < 3; ++j) {
    const int c = tid + 256 * j;
    float fus = ev[j] * g0 + xv3[j] * g1;
    fus = fus * (1.f - mi) + mask_emb[c] * mi;
    const float base = (c < 384) ? yv : xv;
    const int jj = (c < 384) ? c : (c - 384);
    const int m = jj >> 1;
    const float inv_dt = __expf(-((float)m / 192.f) * LN10000_F);
    const float ph = base * inv_dt;
    fus += (jj & 1) ? __cosf(ph) : __sinf(ph);
    vals[j] = fus; sum += fus; sq += fus * fus;
  }
  __syncthreads();
  r0[tid] = sum; r1[tid] = sq; __syncthreads();
  for (int s = 128; s > 0; s >>= 1) {
    if (tid < s) { r0[tid] += r0[tid + s]; r1[tid] += r1[tid + s]; }
    __syncthreads();
  }
  const float mu = r0[0] * (1.f / 768.f);
  const float var = r1[0] * (1.f / 768.f) - mu * mu;
  const float inv = rsqrtf(var + 1e-5f);
  for (int j = 0; j < 3; ++j) {
    const int c = tid + 256 * j;
    out[(size_t)t * 768 + c] = (vals[j] - mu) * inv * ln_g[c] + ln_b[c];
  }
  if (tid == 0) {
    vmOut[t]  = vmv;
    labOut[t] = (mi > 0.f) ? indv : -100.f;
  }
}

// ---------------------------------------------------------------------------
// Launch
// ---------------------------------------------------------------------------
extern "C" void kernel_launch(void* const* d_in, const int* in_sizes, int n_in,
                              void* d_out, int out_size, void* d_ws, size_t ws_size,
                              hipStream_t stream) {
  (void)in_sizes; (void)n_in; (void)out_size; (void)ws_size;
  const float* img      = (const float*)d_in[0];
  const int*   vis_hw   = (const int*)  d_in[1];
  const int*   tmp_pos  = (const int*)  d_in[2];
  const float* bern     = (const float*)d_in[3];
  const float* conv_w   = (const float*)d_in[4];
  const float* conv_b   = (const float*)d_in[5];
  const float* codebook = (const float*)d_in[6];
  const float* pos_w    = (const float*)d_in[7];
  const float* pos_b    = (const float*)d_in[8];
  const float* gate_w   = (const float*)d_in[9];
  const float* gate_b   = (const float*)d_in[10];
  const float* ln_g     = (const float*)d_in[11];
  const float* ln_b     = (const float*)d_in[12];
  const float* mask_emb = (const float*)d_in[13];

  const int M = 16384;                 // tokens
  // workspace carve-up
  char* ws = (char*)d_ws;
  unsigned short* xt_bf = (unsigned short*)ws;            ws += (size_t)M * 2048 * 2;
  float*          xq_f  = (float*)ws;                     ws += (size_t)M * 768 * 4;
  unsigned short* xq_bf = (unsigned short*)ws;            ws += (size_t)M * 768 * 2;
  float*          emb_f = (float*)ws;                     ws += (size_t)M * 768 * 4;
  unsigned short* cw_bf = (unsigned short*)ws;            ws += (size_t)768 * 2048 * 2;
  unsigned short* cb_bf = (unsigned short*)ws;            ws += (size_t)2048 * 768 * 2;
  unsigned short* pw_bf = (unsigned short*)ws;            ws += (size_t)768 * 768 * 2;
  float*          hn    = (float*)ws;                     ws += (size_t)2048 * 4;
  int*            idx   = (int*)ws;                       ws += (size_t)M * 4;

  float* out_main = (float*)d_out;
  float* vm_out   = out_main + (size_t)M * 768;
  float* lab_out  = vm_out + M;

  // 1) maxpool + pack A (bf16, token-major)
  maxpool_pack_kernel<<<16384, 256, 0, stream>>>(img, xt_bf);

  // 2) weight conversions
  cvt_bf16_kernel<<<(768 * 2048 + 255) / 256, 256, 0, stream>>>(conv_w, cw_bf, 768 * 2048);
  cvt_bf16_kernel<<<(2048 * 768 + 255) / 256, 256, 0, stream>>>(codebook, cb_bf, 2048 * 768);
  cvt_bf16_kernel<<<(768 * 768 + 255) / 256, 256, 0, stream>>>(pos_w, pw_bf, 768 * 768);

  // 3) codebook half-norms
  halfnorm_kernel<<<2048 / 8, 256, 0, stream>>>(codebook, hn);

  // 4) conv GEMM: xq = xt @ conv_w^T + conv_b   (f32 + bf16 copies)
  wmma_gemm_kernel<<<dim3(M / 64, 768 / 256), 256, 0, stream>>>(
      xt_bf, nullptr, cw_bf, conv_b, xq_f, xq_bf, 768, 2048);

  // 5) VQ: idx = argmin_k ||xq - codebook_k||^2
  vq_argmax_kernel<<<M / 64, 256, 0, stream>>>(xq_bf, cb_bf, hn, idx, 768, 2048);

  // 6) emb = codebook[idx] @ pos_w^T + pos_b  (A gathered through idx)
  wmma_gemm_kernel<<<dim3(M / 64, 768 / 256), 256, 0, stream>>>(
      cb_bf, idx, pw_bf, pos_b, emb_f, nullptr, 768, 768);

  // 7) fused epilogue
  epilogue_kernel<<<M, 256, 0, stream>>>(emb_f, xq_f, idx, vis_hw, tmp_pos, bern,
                                         gate_w, gate_b, mask_emb, ln_g, ln_b,
                                         out_main, vm_out, lab_out);
}
